// MultiHeadDotProductAttention_24756191494231
// MI455X (gfx1250) — matrix operand order repaired and checked by reading, not compiled, not run on hardware
//
#include <hip/hip_runtime.h>

#define NBX 24
#define NBY 24
#define BATCH 8
#define NH 12
#define DH 64
#define SEQ 577
#define GRID_TOK 576
#define NPAR (4 * NBX * NBY)   // 2304
#define EPSV 1e-8f

typedef __attribute__((ext_vector_type(16))) _Float16 v16h;
typedef __attribute__((ext_vector_type(8)))  float    v8f;

union V16U { uint4 u[2]; v16h h; };

// 16 contiguous f16 from LDS (16B-aligned by construction) -> two b128 loads
__device__ __forceinline__ v16h lds_ld16(const _Float16* p) {
  V16U r;
  r.u[0] = *(const uint4*)(const void*)(p);
  r.u[1] = *(const uint4*)(const void*)(p + 8);
  return r.h;
}

// 8 floats (two float4) -> relu(x*scale)+eps -> f16, into a[base..base+7]
__device__ __forceinline__ void qput8(v16h& a, int base, const float* p, float scale) {
  float4 x = *(const float4*)(const void*)(p);
  float4 y = *(const float4*)(const void*)(p + 4);
  a[base + 0] = (_Float16)(fmaxf(x.x * scale, 0.f) + EPSV);
  a[base + 1] = (_Float16)(fmaxf(x.y * scale, 0.f) + EPSV);
  a[base + 2] = (_Float16)(fmaxf(x.z * scale, 0.f) + EPSV);
  a[base + 3] = (_Float16)(fmaxf(x.w * scale, 0.f) + EPSV);
  a[base + 4] = (_Float16)(fmaxf(y.x * scale, 0.f) + EPSV);
  a[base + 5] = (_Float16)(fmaxf(y.y * scale, 0.f) + EPSV);
  a[base + 6] = (_Float16)(fmaxf(y.z * scale, 0.f) + EPSV);
  a[base + 7] = (_Float16)(fmaxf(y.w * scale, 0.f) + EPSV);
}

#define WMMA_F16(A, B, C) \
  __builtin_amdgcn_wmma_f32_16x16x32_f16(false, (A), false, (B), (short)0, (C), false, false)

__global__ __launch_bounds__(256)
void topo_linear_attn(const float* __restrict__ Q,
                      const float* __restrict__ K,
                      const float* __restrict__ V,
                      const float* __restrict__ P,
                      float* __restrict__ O) {
  // Padded strides: Klds row = 144B, Vtlds row = 80B, Slds row = 80B (16B aligned, bank-spread)
  __shared__ _Float16 Klds[32][72];     // [key][dim]   (keys x 64 dims, f16 features)
  __shared__ _Float16 Vtlds[64][40];    // [dim][key]   (transposed V tile)
  __shared__ _Float16 Slds[8][16][40];  // per-wave masked scores (16 q x 32 k)
  __shared__ float    Plds[NPAR];       // |topological_params[h]|

  const int tid  = threadIdx.x;
  const int lane = tid & 31;
  const int wid  = tid >> 5;
  const int m    = lane & 15;
  const int hi   = lane >> 4;

  const int b     = blockIdx.z;
  const int h     = blockIdx.y;
  const int qbase = blockIdx.x * 128;

  // stage |params| for this head
  for (int i = tid; i < NPAR; i += 256)
    Plds[i] = fabsf(P[(size_t)h * NPAR + i]);

  // ---- Q tile -> registers, WMMA A layout (row m = lane&15) ----
  const int qrowA = qbase + wid * 16 + m;
  const int qcl   = qrowA < SEQ ? qrowA : SEQ - 1;
  const float* qp = Q + (((size_t)b * SEQ + qcl) * NH + h) * DH;
  v16h aq0, aq1;
  qput8(aq0, 0, qp + hi * 16,      0.125f);   // d = hi*16 .. +7
  qput8(aq0, 8, qp + hi * 16 + 8, 0.125f);   // d = hi*16+8 .. +7
  qput8(aq1, 0, qp + 32 + hi * 16, 0.125f);
  qput8(aq1, 8, qp + 40 + hi * 16, 0.125f);

  // Toeplitz constants for the 8 C-layout rows this lane touches
  const int qrow0 = qbase + wid * 16 + 8 * hi;
  int qc[8];
#pragma unroll
  for (int j = 0; j < 8; ++j) {
    int tq = qrow0 + j - 1;
    tq = tq < 0 ? 0 : (tq > GRID_TOK - 1 ? GRID_TOK - 1 : tq);
    qc[j] = (tq / NBY + NBX) * (2 * NBY) + (tq % NBY) + NBY;
  }

  const v8f vzero = {0.f, 0.f, 0.f, 0.f, 0.f, 0.f, 0.f, 0.f};
  v8f acc0 = vzero, acc1 = vzero, acc2 = vzero, acc3 = vzero;
  float rs = 0.f;

  const int r = tid >> 3;   // staging: key row 0..31
  const int g = tid & 7;    // staging: dim group (8 dims each)
  const int NT = (SEQ + 31) / 32;   // 19 key tiles

  for (int kt = 0; kt < NT; ++kt) {
    const int kbase = kt * 32;
    __syncthreads();
    // ---- stage K (relu features) and V (transposed) ----
    {
      const int kr = kbase + r;
      const bool kv = kr < SEQ;
      const size_t off = (((size_t)b * SEQ + (kv ? kr : 0)) * NH + h) * DH + g * 8;
      float4 k0 = *(const float4*)(const void*)(K + off);
      float4 k1 = *(const float4*)(const void*)(K + off + 4);
      float4 v0 = *(const float4*)(const void*)(V + off);
      float4 v1 = *(const float4*)(const void*)(V + off + 4);
      _Float16* kd = &Klds[r][g * 8];
      kd[0] = kv ? (_Float16)(fmaxf(k0.x, 0.f) + EPSV) : (_Float16)0.f;
      kd[1] = kv ? (_Float16)(fmaxf(k0.y, 0.f) + EPSV) : (_Float16)0.f;
      kd[2] = kv ? (_Float16)(fmaxf(k0.z, 0.f) + EPSV) : (_Float16)0.f;
      kd[3] = kv ? (_Float16)(fmaxf(k0.w, 0.f) + EPSV) : (_Float16)0.f;
      kd[4] = kv ? (_Float16)(fmaxf(k1.x, 0.f) + EPSV) : (_Float16)0.f;
      kd[5] = kv ? (_Float16)(fmaxf(k1.y, 0.f) + EPSV) : (_Float16)0.f;
      kd[6] = kv ? (_Float16)(fmaxf(k1.z, 0.f) + EPSV) : (_Float16)0.f;
      kd[7] = kv ? (_Float16)(fmaxf(k1.w, 0.f) + EPSV) : (_Float16)0.f;
      const int db = g * 8;
      Vtlds[db + 0][r] = kv ? (_Float16)v0.x : (_Float16)0.f;
      Vtlds[db + 1][r] = kv ? (_Float16)v0.y : (_Float16)0.f;
      Vtlds[db + 2][r] = kv ? (_Float16)v0.z : (_Float16)0.f;
      Vtlds[db + 3][r] = kv ? (_Float16)v0.w : (_Float16)0.f;
      Vtlds[db + 4][r] = kv ? (_Float16)v1.x : (_Float16)0.f;
      Vtlds[db + 5][r] = kv ? (_Float16)v1.y : (_Float16)0.f;
      Vtlds[db + 6][r] = kv ? (_Float16)v1.z : (_Float16)0.f;
      Vtlds[db + 7][r] = kv ? (_Float16)v1.w : (_Float16)0.f;
    }
    __syncthreads();

    // ---- GEMM1: S[16q x 32k] = Qf @ Kf^T  (B layout: lane N=key, K=dims) ----
    v8f s0 = vzero, s1 = vzero;
    {
      v16h b00 = lds_ld16(&Klds[m][hi * 16]);            // ksub0, d 0..31
      v16h b01 = lds_ld16(&Klds[m][32 + hi * 16]);       // ksub0, d 32..63
      v16h b10 = lds_ld16(&Klds[16 + m][hi * 16]);       // ksub1, d 0..31
      v16h b11 = lds_ld16(&Klds[16 + m][32 + hi * 16]);  // ksub1, d 32..63
      s0 = WMMA_F16(aq0, b00, s0);
      s0 = WMMA_F16(aq1, b01, s0);
      s1 = WMMA_F16(aq0, b10, s1);
      s1 = WMMA_F16(aq1, b11, s1);
    }

    // ---- mask: s = s*|M| + eps (0 for padded keys) ----
#pragma unroll
    for (int t = 0; t < 2; ++t) {
      v8f& s = t ? s1 : s0;
      const int ki = kbase + t * 16 + m;
      const bool kvalid = ki < SEQ;
      const bool kcls = (ki == 0);
      const int tk = kcls ? 0 : ki - 1;
      const int kdelta = (tk / NBY) * (2 * NBY) + (tk % NBY);
#pragma unroll
      for (int j = 0; j < 8; ++j) {
        const int qi = qrow0 + j;
        const float w = (kcls || qi == 0) ? 1.0f : Plds[qc[j] - kdelta];
        const float val = fmaf(s[j], w, EPSV);
        s[j] = kvalid ? val : 0.0f;
      }
    }

    // ---- C layout -> A layout via per-wave LDS bounce ----
#pragma unroll
    for (int j = 0; j < 8; ++j) {
      Slds[wid][j + 8 * hi][m]      = (_Float16)s0[j];
      Slds[wid][j + 8 * hi][16 + m] = (_Float16)s1[j];
    }
    V16U su;
    su.u[0] = *(const uint4*)(const void*)&Slds[wid][m][hi * 16];
    su.u[1] = *(const uint4*)(const void*)&Slds[wid][m][hi * 16 + 8];
    v16h sa = su.h;

    // partial row sum (this lane holds half the 32 keys of row m)
    float part = 0.f;
#pragma unroll
    for (int i = 0; i < 16; ++i) part += (float)sa[i];
    rs += part;

    // ---- GEMM2: O[16q x 64d] += S @ V  (B layout: lane N=dim, K=keys) ----
    v16h bv0 = lds_ld16(&Vtlds[m][hi * 16]);
    v16h bv1 = lds_ld16(&Vtlds[16 + m][hi * 16]);
    v16h bv2 = lds_ld16(&Vtlds[32 + m][hi * 16]);
    v16h bv3 = lds_ld16(&Vtlds[48 + m][hi * 16]);
    acc0 = WMMA_F16(sa, bv0, acc0);
    acc1 = WMMA_F16(sa, bv1, acc1);
    acc2 = WMMA_F16(sa, bv2, acc2);
    acc3 = WMMA_F16(sa, bv3, acc3);
  }

  // ---- normalize and store ----
  const float rtot = rs + __shfl_xor(rs, 16, 32);  // full row sum for row m (both half-lanes)
#pragma unroll
  for (int j = 0; j < 8; ++j) {
    const float rj = __shfl(rtot, 8 * hi + j, 32); // rowsum for C-layout row j+8*hi
    const float inv = 1.0f / (rj + (float)SEQ * EPSV);
    const int qi = qrow0 + j;
    if (qi < SEQ) {
      float* op = O + (((size_t)b * SEQ + qi) * NH + h) * DH + m;
      op[0]  = acc0[j] * inv;
      op[16] = acc1[j] * inv;
      op[32] = acc2[j] * inv;
      op[48] = acc3[j] * inv;
    }
  }
}

extern "C" void kernel_launch(void* const* d_in, const int* in_sizes, int n_in,
                              void* d_out, int out_size, void* d_ws, size_t ws_size,
                              hipStream_t stream) {
  const float* Q = (const float*)d_in[0];
  const float* K = (const float*)d_in[1];
  const float* V = (const float*)d_in[2];
  const float* P = (const float*)d_in[3];
  float* O = (float*)d_out;
  dim3 grid((SEQ + 127) / 128, NH, BATCH);   // 5 x 12 x 8 = 480 workgroups
  topo_linear_attn<<<grid, dim3(256), 0, stream>>>(Q, K, V, P, O);
}
